// MultiHeadAttention_60541859004769
// MI455X (gfx1250) — matrix operand order repaired and checked by reading, not compiled, not run on hardware
//
#include <hip/hip_runtime.h>
#include <hip/hip_bf16.h>

// ---------------------------------------------------------------------------
// MI455X (gfx1250) multi-head attention block, bf16 WMMA + f32 accumulate.
//   B=16, LV=LS=512, FEA_V=2048, FEA_S=768, D=1024, H=16, DK=DV=64
// GEMMs: TDM (tensor_load_to_lds) stages 64x64 bf16 weight tiles into LDS,
// double-buffered, shared by 8 waves; each wave computes a 32x64 WMMA tile.
// ---------------------------------------------------------------------------

#define B_    16
#define LV_   512
#define LS_   512
#define FEAV  2048
#define FEAS  768
#define DM    1024
#define H_    16
#define DHEAD 64
#define NTOK  (B_ * LV_)        // 8192 rows (same for s side)

typedef __attribute__((ext_vector_type(16))) __bf16 v16bf;
typedef __attribute__((ext_vector_type(8)))  float  v8f;
typedef unsigned int u32x4 __attribute__((ext_vector_type(4)));
typedef int          i32x8 __attribute__((ext_vector_type(8)));
typedef int          i32x4 __attribute__((ext_vector_type(4)));

__device__ __forceinline__ unsigned short f2bf(float f) {
    unsigned u = __float_as_uint(f);
    unsigned r = u + 0x7FFFu + ((u >> 16) & 1u);   // round-to-nearest-even
    return (unsigned short)(r >> 16);
}

union FragU { uint4 u[2]; v16bf v; };

__device__ __forceinline__ v16bf mk_frag(const unsigned short* p0,
                                         const unsigned short* p1) {
    FragU t;
    t.u[0] = *(const uint4*)p0;   // 8 bf16
    t.u[1] = *(const uint4*)p1;   // 8 bf16
    return t.v;
}

__device__ __forceinline__ v8f wmma_bf16(v16bf a, v16bf b, v8f c) {
    return __builtin_amdgcn_wmma_f32_16x16x32_bf16(
        /*neg_a=*/false, a, /*neg_b=*/false, b,
        /*c_mod=*/(short)0, c, /*reuse_a=*/false, /*reuse_b=*/false);
}

// Tensor Data Mover: DMA a 64x64 bf16 tile (row stride = K elements) from
// global memory into LDS at byte offset lds_off. D# layout per CDNA5 ISA
// CDNA5 ISA (group0: count/lds/global/type, group1: dims/strides).
// 6-arg builtin form (clang-23 / therock-10.0 toolchain).
__device__ __forceinline__ void tdm_load_tile(const unsigned short* gsrc,
                                              unsigned lds_off,
                                              unsigned K, unsigned Nrows) {
    unsigned long long ga = (unsigned long long)(uintptr_t)gsrc;
    u32x4 g0;
    g0[0] = 1u;                                     // count=1, no gather
    g0[1] = lds_off;                                // LDS byte address
    g0[2] = (unsigned)ga;                           // global_addr[31:0]
    g0[3] = (unsigned)((ga >> 32) & 0x01FFFFFFu)    // global_addr[56:32]
          | 0x80000000u;                            // type=2 ("image")
    i32x8 g1;
    g1[0] = 0x10000;                                // data_size=1 (2 bytes)
    g1[1] = (int)((K & 0xFFFFu) << 16);             // tensor_dim0 lo
    g1[2] = (int)((K >> 16) & 0xFFFFu)              // tensor_dim0 hi
          | (int)((Nrows & 0xFFFFu) << 16);         // tensor_dim1 lo
    g1[3] = (int)((Nrows >> 16) & 0xFFFFu)          // tensor_dim1 hi
          | (64 << 16);                             // tile_dim0 = 64
    g1[4] = 64;                                     // tile_dim1=64, tile_dim2=0
    g1[5] = (int)K;                                 // tensor_dim0_stride lo
    g1[6] = 0;                                      // stride hi / dim1_stride lo
    g1[7] = 0;
    i32x4 z4 = {0, 0, 0, 0};                        // groups 2/3 unused (2D)
    i32x8 z8 = {0, 0, 0, 0, 0, 0, 0, 0};
    __builtin_amdgcn_tensor_load_to_lds(g0, g1, z4, z4, z8, 0);
}

// ---------------------------------------------------------------------------
// Elementwise conversion kernels
// ---------------------------------------------------------------------------
__global__ void cvt_bf16(const float* __restrict__ src,
                         unsigned short* __restrict__ dst, int n) {
    int i = blockIdx.x * blockDim.x + threadIdx.x;
    if (i < n) dst[i] = f2bf(src[i]);
}

// W[K,N] f32 (row major)  ->  Wt[N,K] bf16 (row major): B transposed so the
// WMMA B-operand (lane n = column, contiguous K) loads contiguously.
__global__ void cvt_transpose_bf16(const float* __restrict__ W,
                                   unsigned short* __restrict__ Wt,
                                   int K, int N) {
    int i = blockIdx.x * blockDim.x + threadIdx.x;
    if (i < K * N) {
        int k = i / N, n = i % N;
        Wt[(size_t)n * K + k] = f2bf(W[i]);
    }
}

// VS[b, key, h*64+d] bf16 -> VST[((b*16+h)*64+d)*512 + key] bf16
__global__ void transpose_vs(const unsigned short* __restrict__ VS,
                             unsigned short* __restrict__ VST) {
    int i = blockIdx.x * blockDim.x + threadIdx.x;
    if (i < B_ * H_ * DHEAD * LS_) {
        int key = i & 511;
        int d   = (i >> 9) & 63;
        int h   = (i >> 15) & 15;
        int b   = i >> 19;
        VST[i] = VS[(size_t)(b * LS_ + key) * DM + h * DHEAD + d];
    }
}

// ---------------------------------------------------------------------------
// bf16 WMMA GEMM: C[M,N] = A[M,K] x Bt[N,K]^T (+ bias, epilogue)
// Block: 8 waves, 256x64 output tile. B tile (64xK) staged via TDM into LDS,
// double-buffered; each wave computes 32x64 (2 M-subtiles x 4 N-subtiles).
// modes: 0 f32 store | 1 relu->bf16 | 2 relu->bf16 + f32 (val+pos) residual
//        3 scale->bf16 | 4 f32 (val+residual)
// ---------------------------------------------------------------------------
__global__ __launch_bounds__(256)
void gemm_wmma(const unsigned short* __restrict__ A,
               const unsigned short* __restrict__ Bt,
               const float* __restrict__ bias,
               unsigned short* __restrict__ Cbf,
               float* __restrict__ Cf,
               const float* __restrict__ aux,
               int M, int N, int K, int mode, float scale, int posMod) {
    __shared__ __align__(16) unsigned short ldsB[2][64 * 64];   // 2 x 8KB

    int tilesN = N >> 6;
    int bm = blockIdx.x / tilesN;
    int bn = blockIdx.x % tilesN;
    int n0 = bn << 6;
    int wave = threadIdx.x >> 5;
    int m0 = (bm << 8) + (wave << 5);          // 32 rows per wave
    int lane = threadIdx.x & 31;
    int lr = lane & 15, lh = lane >> 4;

    v8f acc0[4] = {}, acc1[4] = {};
    const unsigned short* arow0 = A + (size_t)(m0 + lr) * K;
    const unsigned short* arow1 = A + (size_t)(m0 + 16 + lr) * K;
    const unsigned short* btile = Bt + (size_t)n0 * K;

    unsigned ldsOff0 = (unsigned)(uintptr_t)(const void*)&ldsB[0][0];
    unsigned ldsOff1 = (unsigned)(uintptr_t)(const void*)&ldsB[1][0];

    int nChunks = K >> 6;                       // 64-wide k chunks
    if (wave == 0)
        tdm_load_tile(btile, ldsOff0, (unsigned)K, (unsigned)N);

    for (int kc = 0; kc < nChunks; ++kc) {
        if (wave == 0) {
            if (kc + 1 < nChunks) {
                tdm_load_tile(btile + (kc + 1) * 64,
                              ((kc + 1) & 1) ? ldsOff1 : ldsOff0,
                              (unsigned)K, (unsigned)N);
                __builtin_amdgcn_s_wait_tensorcnt(1);   // current chunk ready
            } else {
                __builtin_amdgcn_s_wait_tensorcnt(0);
            }
        }
        __syncthreads();

        const unsigned short* bbase = &ldsB[kc & 1][0];
        int kbase = kc << 6;
        if (kc + 2 < nChunks) {                 // prefetch A stream
            __builtin_prefetch(arow0 + kbase + 128, 0, 3);
            __builtin_prefetch(arow1 + kbase + 128, 0, 3);
        }
#pragma unroll
        for (int ks = 0; ks < 64; ks += 32) {
            int ka = kbase + ks + (lh << 3);    // A: +8 cols for hi lanes
            v16bf af0 = mk_frag(arow0 + ka, arow0 + ka + 16);
            v16bf af1 = mk_frag(arow1 + ka, arow1 + ka + 16);
            int kb = ks + (lh << 3);            // B: +8 rows for hi lanes
#pragma unroll
            for (int t = 0; t < 4; ++t) {
                const unsigned short* bl = bbase + (t * 16 + lr) * 64 + kb;
                v16bf bf = mk_frag(bl, bl + 16);
                acc0[t] = wmma_bf16(af0, bf, acc0[t]);
                acc1[t] = wmma_bf16(af1, bf, acc1[t]);
            }
        }
        __syncthreads();                        // before buffer reuse
    }

#pragma unroll
    for (int half = 0; half < 2; ++half) {
        v8f* acc = half ? acc1 : acc0;
        int mb = m0 + half * 16;
#pragma unroll
        for (int t = 0; t < 4; ++t) {
            int n = n0 + t * 16 + lr;
            float bv = bias ? bias[n] : 0.0f;
#pragma unroll
            for (int r = 0; r < 8; ++r) {
                int m = mb + r + (lh << 3);
                float v = acc[t][r] + bv;
                size_t idx = (size_t)m * N + n;
                if (mode == 0) {
                    Cf[idx] = v;
                } else if (mode == 1) {
                    v = v > 0.0f ? v : 0.0f;
                    Cbf[idx] = f2bf(v);
                } else if (mode == 2) {
                    v = v > 0.0f ? v : 0.0f;
                    Cbf[idx] = f2bf(v);
                    Cf[idx] = v + aux[(size_t)(m % posMod) * N + n];
                } else if (mode == 3) {
                    Cbf[idx] = f2bf(v * scale);
                } else {
                    Cf[idx] = v + aux[idx];
                }
            }
        }
    }
}

// ---------------------------------------------------------------------------
// Fused attention: one wave per (b, h, 16-row q tile).
//  scores = Q(scaled) x VS^T via WMMA -> LDS f32 [16][512]
//  exact softmax per row (lanes 0..15) -> P bf16 in LDS
//  context = P x VS via WMMA (VST pre-transposed [b,h,d,key])
// ---------------------------------------------------------------------------
__global__ __launch_bounds__(32)
void attn_wmma(const unsigned short* __restrict__ Q,
               const unsigned short* __restrict__ VS,
               const unsigned short* __restrict__ VST,
               unsigned short* __restrict__ CTX) {
    __shared__ __align__(16) float          sc[16][LS_];
    __shared__ __align__(16) unsigned short pb[16][LS_];
    __shared__ float rsum[16];

    int id = blockIdx.x;                 // b*H*32 + h*32 + qt
    int qt = id & 31;
    int h  = (id >> 5) & 15;
    int b  = id >> 9;
    int m0 = qt << 4;
    int lane = threadIdx.x & 31;
    int lr = lane & 15, lh = lane >> 4;

    // Q A-fragments for DK=64 (two k-steps of 32); 1/sqrt(DK) folded into Q.
    const unsigned short* qrow =
        Q + (size_t)(b * LV_ + m0 + lr) * DM + h * DHEAD;
    int ka = lh << 3;
    v16bf aq0 = mk_frag(qrow + ka,      qrow + ka + 16);
    v16bf aq1 = mk_frag(qrow + 32 + ka, qrow + 32 + ka + 16);

    // Pass 1: scores -> LDS
    for (int n0 = 0; n0 < LS_; n0 += 16) {
        const unsigned short* krow =
            VS + (size_t)(b * LS_ + n0 + lr) * DM + h * DHEAD;
        int kb = lh << 3;
        v16bf b0 = mk_frag(krow + kb,      krow + kb + 16);
        v16bf b1 = mk_frag(krow + 32 + kb, krow + 32 + kb + 16);
        v8f s = {};
        s = wmma_bf16(aq0, b0, s);
        s = wmma_bf16(aq1, b1, s);
#pragma unroll
        for (int r = 0; r < 8; ++r)
            sc[r + (lh << 3)][n0 + lr] = s[r];
    }
    __syncthreads();

    // Pass 2: exact softmax per row; store P as bf16.
    if (lane < 16) {
        float mx = -1e30f;
        for (int j = 0; j < LS_; ++j) mx = fmaxf(mx, sc[lane][j]);
        float sum = 0.0f;
        for (int j = 0; j < LS_; ++j) {
            float e = __expf(sc[lane][j] - mx);
            sum += e;
            pb[lane][j] = f2bf(e);
        }
        rsum[lane] = sum;
    }
    __syncthreads();

    // Pass 3: context = P x VS   (16 x 512) x (512 x 64)
    v8f acc[4] = {};
    for (int k0 = 0; k0 < LS_; k0 += 32) {
        int kpa = k0 + (lh << 3);
        v16bf af = mk_frag(&pb[lr][kpa], &pb[lr][kpa + 16]);
        int kpb = k0 + (lh << 3);
#pragma unroll
        for (int t = 0; t < 4; ++t) {
            const unsigned short* vrow =
                VST + (size_t)((b * H_ + h) * DHEAD + t * 16 + lr) * LS_ + kpb;
            v16bf bf = mk_frag(vrow, vrow + 16);
            acc[t] = wmma_bf16(af, bf, acc[t]);
        }
    }

#pragma unroll
    for (int t = 0; t < 4; ++t) {
#pragma unroll
        for (int r = 0; r < 8; ++r) {
            int rr = r + (lh << 3);
            float v = acc[t][r] / rsum[rr];
            CTX[(size_t)(b * LV_ + m0 + rr) * DM + h * DHEAD + t * 16 + lr] =
                f2bf(v);
        }
    }
}

// ---------------------------------------------------------------------------
// LayerNorm: one wave per 1024-element row.
// ---------------------------------------------------------------------------
__global__ __launch_bounds__(256)
void ln_kernel(const float* __restrict__ Y,
               const float* __restrict__ gamma,
               const float* __restrict__ beta,
               float* __restrict__ out, int rows) {
    int wave = blockIdx.x * (blockDim.x >> 5) + (threadIdx.x >> 5);
    if (wave >= rows) return;
    int lane = threadIdx.x & 31;
    const float* row = Y + (size_t)wave * DM;
    float vals[32];
    float s = 0.0f, s2 = 0.0f;
#pragma unroll
    for (int i = 0; i < 32; ++i) {
        float x = row[lane + i * 32];
        vals[i] = x;
        s += x;
        s2 += x * x;
    }
#pragma unroll
    for (int off = 16; off > 0; off >>= 1) {
        s  += __shfl_xor(s, off, 32);
        s2 += __shfl_xor(s2, off, 32);
    }
    float mu  = s * (1.0f / DM);
    float var = s2 * (1.0f / DM) - mu * mu;
    float inv = rsqrtf(var + 1e-6f);
    float* orow = out + (size_t)wave * DM;
#pragma unroll
    for (int i = 0; i < 32; ++i) {
        int c = lane + i * 32;
        orow[c] = gamma[c] * (vals[i] - mu) * inv + beta[c];
    }
}

// ---------------------------------------------------------------------------
// Host-side orchestration
// ---------------------------------------------------------------------------
extern "C" void kernel_launch(void* const* d_in, const int* in_sizes, int n_in,
                              void* d_out, int out_size, void* d_ws, size_t ws_size,
                              hipStream_t stream) {
    const float* v     = (const float*)d_in[0];
    const float* s     = (const float*)d_in[1];
    const float* W_lv  = (const float*)d_in[2];
    const float* b_lv  = (const float*)d_in[3];
    const float* W_ls  = (const float*)d_in[4];
    const float* b_ls  = (const float*)d_in[5];
    const float* pos_v = (const float*)d_in[6];
    // d_in[7] pos_s: unused; d_in[10]/[11] W_k/b_k: computed-then-discarded in ref
    const float* W_q   = (const float*)d_in[8];
    const float* b_q   = (const float*)d_in[9];
    const float* W_v   = (const float*)d_in[12];
    const float* b_v   = (const float*)d_in[13];
    const float* W_p   = (const float*)d_in[14];
    const float* b_p   = (const float*)d_in[15];
    const float* gamma = (const float*)d_in[16];
    const float* beta  = (const float*)d_in[17];
    float* out = (float*)d_out;

    char* ws = (char*)d_ws;
    // Region A [0, 33.5MB): vbf -> later (qbf | vsbf) -> later ybuf (f32)
    const size_t OFF_A = 0;
    unsigned short* vbf   = (unsigned short*)(ws + OFF_A);
    unsigned short* qbf   = (unsigned short*)(ws + OFF_A);
    unsigned short* vsbf  = (unsigned short*)(ws + OFF_A + (size_t)NTOK * DM * 2);
    float*          ybuf  = (float*)(ws + OFF_A);
    // Region S: s in bf16
    const size_t OFF_S = (size_t)NTOK * FEAV * 2;                 // 33,554,432
    unsigned short* sbf = (unsigned short*)(ws + OFF_S);
    // Weights (bf16, transposed)
    const size_t OFF_W = OFF_S + (size_t)NTOK * FEAS * 2;         // +12,582,912
    unsigned short* WtLv = (unsigned short*)(ws + OFF_W);
    unsigned short* WtLs = (unsigned short*)(ws + OFF_W + (size_t)FEAV * DM * 2);
    unsigned short* WtQ  = (unsigned short*)((char*)WtLs + (size_t)FEAS * DM * 2);
    unsigned short* WtV  = (unsigned short*)((char*)WtQ + (size_t)DM * DM * 2);
    unsigned short* WtP  = (unsigned short*)((char*)WtV + (size_t)DM * DM * 2);
    // Region C: vproj bf16 -> later VST
    const size_t OFF_C = OFF_W + (size_t)(FEAV + FEAS + 3 * DM) * DM * 2;
    unsigned short* vprojbf = (unsigned short*)(ws + OFF_C);
    unsigned short* vstbf   = (unsigned short*)(ws + OFF_C);
    // Region D: residual f32
    const size_t OFF_D = OFF_C + (size_t)NTOK * DM * 2;
    float* resid = (float*)(ws + OFF_D);
    // Region E: sproj bf16 -> later ctx bf16
    const size_t OFF_E = OFF_D + (size_t)NTOK * DM * 4;
    unsigned short* sprojbf = (unsigned short*)(ws + OFF_E);
    unsigned short* ctxbf   = (unsigned short*)(ws + OFF_E);

    const int TB = 256;
    // --- conversions ---
    {
        int n = NTOK * FEAV;
        cvt_bf16<<<(n + TB - 1) / TB, TB, 0, stream>>>(v, vbf, n);
    }
    {
        int n = NTOK * FEAS;
        cvt_bf16<<<(n + TB - 1) / TB, TB, 0, stream>>>(s, sbf, n);
    }
    cvt_transpose_bf16<<<(FEAV * DM + TB - 1) / TB, TB, 0, stream>>>(W_lv, WtLv, FEAV, DM);
    cvt_transpose_bf16<<<(FEAS * DM + TB - 1) / TB, TB, 0, stream>>>(W_ls, WtLs, FEAS, DM);
    cvt_transpose_bf16<<<(DM * DM + TB - 1) / TB, TB, 0, stream>>>(W_q, WtQ, DM, DM);
    cvt_transpose_bf16<<<(DM * DM + TB - 1) / TB, TB, 0, stream>>>(W_v, WtV, DM, DM);
    cvt_transpose_bf16<<<(DM * DM + TB - 1) / TB, TB, 0, stream>>>(W_p, WtP, DM, DM);

    // --- GEMMs: blocks of 8 waves computing 256x64; grid = (M/256)*(N/64) ---
    const int gemmBlocks = (NTOK / 256) * (DM / 64);   // 32*16 = 512

    // 1) vproj = relu(v @ W_lv + b_lv); residual = vproj + pos_v
    gemm_wmma<<<gemmBlocks, TB, 0, stream>>>(vbf, WtLv, b_lv, vprojbf, resid,
                                             pos_v, NTOK, DM, FEAV, 2, 1.0f, LV_);
    // 2) sproj = relu(s @ W_ls + b_ls)
    gemm_wmma<<<gemmBlocks, TB, 0, stream>>>(sbf, WtLs, b_ls, sprojbf, nullptr,
                                             nullptr, NTOK, DM, FEAS, 1, 1.0f, 1);
    // 3) q = (vproj @ W_q + b_q) * (1/sqrt(DK))  -> bf16
    gemm_wmma<<<gemmBlocks, TB, 0, stream>>>(vprojbf, WtQ, b_q, qbf, nullptr,
                                             nullptr, NTOK, DM, DM, 3, 0.125f, 1);
    // 4) vs = relu(sproj @ W_v + b_v)  -> bf16 (serves as both K and V)
    gemm_wmma<<<gemmBlocks, TB, 0, stream>>>(sprojbf, WtV, b_v, vsbf, nullptr,
                                             nullptr, NTOK, DM, DM, 1, 1.0f, 1);
    // 5) VST[b,h,d,key] = vs transposed per head
    {
        int n = B_ * H_ * DHEAD * LS_;
        transpose_vs<<<(n + TB - 1) / TB, TB, 0, stream>>>(vsbf, vstbf);
    }
    // 6) fused attention: one wave per (b, h, 16-row q tile)
    attn_wmma<<<B_ * H_ * (LV_ / 16), 32, 0, stream>>>(qbf, vsbf, vstbf, ctxbf);
    // 7) y = residual + ctx @ W_p + b_p  (fp32)
    gemm_wmma<<<gemmBlocks, TB, 0, stream>>>(ctxbf, WtP, b_p, nullptr, ybuf,
                                             resid, NTOK, DM, DM, 4, 1.0f, 1);
    // 8) LayerNorm -> out
    ln_kernel<<<(NTOK + 7) / 8, TB, 0, stream>>>(ybuf, gamma, beta, out, NTOK);
}
